// Evoformer_Trunk_8538394985042
// MI455X (gfx1250) — compile-verified, matrix operand order checked by reading, not run on hardware
//
#include <hip/hip_runtime.h>

typedef __attribute__((ext_vector_type(16))) _Float16 v16h;
typedef __attribute__((ext_vector_type(8)))  _Float16 h8;
typedef __attribute__((ext_vector_type(4)))  _Float16 h4;
typedef __attribute__((ext_vector_type(8)))  float    v8f;
typedef __attribute__((ext_vector_type(4)))  float    f4;

#define KSH_NONE 30   // K-decomposition disabled: gk>>30 == 0, gk&(2^30-1) == gk

// ---------------------------------------------------------------------------
// Fully strided, batched WMMA GEMM:  C = act(alpha*(A.B) + bias) * gate + res
// A(row,k) = A[batch*sA + row*ldaR + (k>>kshA)*ldaKhi + (k&kmaskA)*ldaKlo]
// B(k,n)   = B[batch*sB + n*ldbN  + (k>>kshB)*ldbKhi + (k&kmaskB)*ldbKlo]
// C(row,n) = C[batch*sC + row*ldcR + n*ldcN]   (gate/res share C strides)
// f32 in/out, f16 WMMA compute, f32 accumulate.  REQUIRES K % 32 == 0.
// ---------------------------------------------------------------------------
struct GemmP {
  const float* A; const float* B; float* C;
  const float* bias; const float* residual; const float* gate;
  int M, N, K, act;                    // act: 0=none 1=relu 2=sigmoid
  int kshA, kmaskA, kshB, kmaskB;     // power-of-two K decomposition
  long long ldaR, ldaKlo, ldaKhi;
  long long ldbN, ldbKlo, ldbKhi;
  long long ldcR, ldcN;
  long long sA, sB, sC;
  float alpha;
};

__launch_bounds__(128)
__global__ void gemm_f16_wmma_kernel(GemmP p) {
  // 40-half row pitch = 80B: 16B-aligned fragment bases, bank-conflict-free
  __shared__ alignas(16) _Float16 Ash[64][40];
  __shared__ alignas(16) _Float16 Bsh[64][40];
  const int tid  = threadIdx.x;
  const int lane = tid & 31;
  const int wave = tid >> 5;
  const int bm = blockIdx.y * 64;
  const int bn = blockIdx.x * 64;
  const long long batch = blockIdx.z;
  const float* A = p.A + batch * p.sA;
  const float* B = p.B + batch * p.sB;
  const int wm = (wave >> 1) * 32;
  const int wn = (wave & 1) * 32;
  const int lrow = lane & 15;
  const int colbase = (lane >> 4) * 8;   // K-half base per ISA 16-bit A/B layout

  // Block-uniform path selection (SCC branches, no per-lane predication).
  const bool fullM = (bm + 64 <= p.M);
  const bool fullN = (bn + 64 <= p.N);
  // 4-wide K chunks start at k%4==0 so they never cross the 32-elem gather
  // group; strides multiple of 4 floats keep 16B alignment.
  const bool avec  = fullM && p.ldaKlo == 1 && (p.ldaR & 3) == 0 &&
                     (p.ldaKhi & 3) == 0;
  const bool bveck = fullN && p.ldbKlo == 1 && (p.ldbN & 3) == 0 &&
                     (p.ldbKhi & 3) == 0;
  const bool bvecn = fullN && p.ldbN == 1 && (p.ldbKlo & 3) == 0 &&
                     (p.ldbKhi & 3) == 0;
  v8f acc[2][2] = {};

  for (int k0 = 0; k0 < p.K; k0 += 32) {
    // ---------------- A tile fill ----------------
    if (avec) {
#pragma unroll
      for (int i = 0; i < 4; ++i) {
        const int idx4 = tid + i * 128;        // 512 float4 chunks
        const int m = idx4 >> 3;               // 64 rows x 8 chunks
        const int kc = (idx4 & 7) << 2;
        const int gk = k0 + kc;
        const long long off = (long long)(bm + m) * p.ldaR +
            (long long)(gk >> p.kshA) * p.ldaKhi + (long long)(gk & p.kmaskA);
        const f4 vv = *(const f4*)(A + off);
        h4 hv;
#pragma unroll
        for (int e = 0; e < 4; ++e) hv[e] = (_Float16)vv[e];
        *(h4*)&Ash[m][kc] = hv;
      }
    } else if (fullM) {
      for (int idx = tid; idx < 2048; idx += 128) {
        const int m = idx >> 5, kk = idx & 31;
        const int gk = k0 + kk;
        Ash[m][kk] = (_Float16)A[(long long)(bm + m) * p.ldaR +
            (long long)(gk >> p.kshA) * p.ldaKhi +
            (long long)(gk & p.kmaskA) * p.ldaKlo];
      }
    } else {
      for (int idx = tid; idx < 2048; idx += 128) {
        const int m = idx >> 5, kk = idx & 31;
        const int gk = k0 + kk;
        float av = 0.f;
        if (bm + m < p.M)
          av = A[(long long)(bm + m) * p.ldaR +
                 (long long)(gk >> p.kshA) * p.ldaKhi +
                 (long long)(gk & p.kmaskA) * p.ldaKlo];
        Ash[m][kk] = (_Float16)av;
      }
    }
    // ---------------- B tile fill ----------------
    if (bveck) {
#pragma unroll
      for (int i = 0; i < 4; ++i) {
        const int idx4 = tid + i * 128;
        const int n = idx4 >> 3;
        const int kc = (idx4 & 7) << 2;
        const int gk = k0 + kc;
        const long long off = (long long)(bn + n) * p.ldbN +
            (long long)(gk >> p.kshB) * p.ldbKhi + (long long)(gk & p.kmaskB);
        const f4 vv = *(const f4*)(B + off);
        h4 hv;
#pragma unroll
        for (int e = 0; e < 4; ++e) hv[e] = (_Float16)vv[e];
        *(h4*)&Bsh[n][kc] = hv;
      }
    } else if (bvecn) {
#pragma unroll
      for (int i = 0; i < 4; ++i) {
        const int idx4 = tid + i * 128;        // 32 k x 16 n-chunks
        const int kk = idx4 >> 4;
        const int nc = (idx4 & 15) << 2;
        const int gk = k0 + kk;
        const long long off = (long long)(gk >> p.kshB) * p.ldbKhi +
            (long long)(gk & p.kmaskB) * p.ldbKlo + (bn + nc);
        const f4 vv = *(const f4*)(B + off);
#pragma unroll
        for (int e = 0; e < 4; ++e) Bsh[nc + e][kk] = (_Float16)vv[e];
      }
    } else if (fullN) {
      for (int idx = tid; idx < 2048; idx += 128) {
        const int n = idx >> 5, kk = idx & 31;
        const int gk = k0 + kk;
        Bsh[n][kk] = (_Float16)B[(long long)(bn + n) * p.ldbN +
            (long long)(gk >> p.kshB) * p.ldbKhi +
            (long long)(gk & p.kmaskB) * p.ldbKlo];
      }
    } else {
      for (int idx = tid; idx < 2048; idx += 128) {
        const int n = idx >> 5, kk = idx & 31;
        const int gk = k0 + kk;
        float bv = 0.f;
        if (bn + n < p.N)
          bv = B[(long long)(bn + n) * p.ldbN +
                 (long long)(gk >> p.kshB) * p.ldbKhi +
                 (long long)(gk & p.kmaskB) * p.ldbKlo];
        Bsh[n][kk] = (_Float16)bv;
      }
    }
    __syncthreads();

    v16h af[2], bf[2];
#pragma unroll
    for (int i = 0; i < 2; ++i) {
      const _Float16* pa = &Ash[wm + i * 16 + lrow][colbase];
      h8 lo = *(const h8*)pa;
      h8 hi = *(const h8*)(pa + 16);
      v16h f;
#pragma unroll
      for (int e = 0; e < 8; ++e) { f[e] = lo[e]; f[e + 8] = hi[e]; }
      af[i] = f;
    }
#pragma unroll
    for (int j = 0; j < 2; ++j) {
      const _Float16* pb = &Bsh[wn + j * 16 + lrow][colbase];
      h8 lo = *(const h8*)pb;
      h8 hi = *(const h8*)(pb + 16);
      v16h f;
#pragma unroll
      for (int e = 0; e < 8; ++e) { f[e] = lo[e]; f[e + 8] = hi[e]; }
      bf[j] = f;
    }
#pragma unroll
    for (int i = 0; i < 2; ++i)
#pragma unroll
      for (int j = 0; j < 2; ++j)
        acc[i][j] = __builtin_amdgcn_wmma_f32_16x16x32_f16(
            false, af[i], false, bf[j], (short)0, acc[i][j], false, false);
    __syncthreads();
  }

  const int lhalf = lane >> 4;
  const bool interior = fullM && fullN;
#pragma unroll
  for (int i = 0; i < 2; ++i)
#pragma unroll
    for (int j = 0; j < 2; ++j)
#pragma unroll
      for (int r = 0; r < 8; ++r) {
        const int gm = bm + wm + i * 16 + r + 8 * lhalf;
        const int gn = bn + wn + j * 16 + lrow;
        if (interior || (gm < p.M && gn < p.N)) {
          float v = acc[i][j][r] * p.alpha;
          if (p.bias) v += p.bias[gn];
          if (p.act == 1) v = fmaxf(v, 0.f);
          else if (p.act == 2) v = 1.f / (1.f + __expf(-v));
          const long long off = batch * p.sC + (long long)gm * p.ldcR +
                                (long long)gn * p.ldcN;
          if (p.gate) v *= p.gate[off];
          if (p.residual) v += p.residual[off];
          p.C[off] = v;
        }
      }
}

// --------------------------- small-N GEMM (N<=4) ---------------------------
struct SmallP {
  const float* A; const float* B; float* C; const float* bias;
  int M, N, K; long long ldaR, ldaK, sA, sC;
};
__global__ void gemm_small_kernel(SmallP p) {
  const int row = blockIdx.x * blockDim.x + threadIdx.x;
  const long long batch = blockIdx.z;
  if (row >= p.M) return;
  const float* A = p.A + batch * p.sA + (long long)row * p.ldaR;
  float acc[4] = {0.f, 0.f, 0.f, 0.f};
  for (int k = 0; k < p.K; ++k) {
    const float a = A[(long long)k * p.ldaK];
    for (int n = 0; n < p.N; ++n) acc[n] += a * p.B[k * p.N + n];
  }
  float* C = p.C + batch * p.sC + (long long)row * p.N;
  for (int n = 0; n < p.N; ++n) C[n] = acc[n] + (p.bias ? p.bias[n] : 0.f);
}

// ------------------------------- LayerNorm ---------------------------------
__global__ void ln_kernel(const float* __restrict__ src, float* __restrict__ dst,
                          const float* g, const float* b, long long rows, int D) {
  const long long row = (long long)blockIdx.x * 8 + (threadIdx.x >> 5);
  const int lane = threadIdx.x & 31;
  if (row >= rows) return;
  const float* x = src + row * D;
  float* o = dst + row * D;
  const int per = D >> 5;               // D in {32,128,256}
  float v[8];
  float s = 0.f;
  for (int i = 0; i < per; ++i) { v[i] = x[lane + (i << 5)]; s += v[i]; }
  for (int off = 16; off; off >>= 1) s += __shfl_xor(s, off, 32);
  const float mu = s / (float)D;
  float q = 0.f;
  for (int i = 0; i < per; ++i) { const float d = v[i] - mu; q += d * d; }
  for (int off = 16; off; off >>= 1) q += __shfl_xor(q, off, 32);
  const float rstd = rsqrtf(q / (float)D + 1e-5f);
  for (int i = 0; i < per; ++i) {
    const int c = lane + (i << 5);
    o[c] = (v[i] - mu) * rstd * g[c] + b[c];
  }
}

// --------------------------- softmax (+head bias) --------------------------
// tSh = log2(T); bias indexed bias[(ti*T+j)*bH + (batch & (bH-1))]
__global__ void softmax_bias_kernel(float* sc, const float* bias,
                                    long long nrows, int tSh, int bH) {
  const long long row = (long long)blockIdx.x * 8 + (threadIdx.x >> 5);
  const int lane = threadIdx.x & 31;
  if (row >= nrows) return;
  const int T = 1 << tSh;
  const long long batch = row >> tSh;
  const int ti = (int)(row & (T - 1));
  float* p = sc + (row << tSh);
  const int per = T >> 5;               // T in {128,256}
  float v[8];
  float mx = -3.4e38f;
  for (int i = 0; i < per; ++i) {
    const int j = lane + (i << 5);
    float x = p[j];
    if (bias) x += bias[(((long long)ti << tSh) + j) * bH + (int)(batch & (bH - 1))];
    v[i] = x; mx = fmaxf(mx, x);
  }
  for (int off = 16; off; off >>= 1) mx = fmaxf(mx, __shfl_xor(mx, off, 32));
  float s = 0.f;
  for (int i = 0; i < per; ++i) { v[i] = __expf(v[i] - mx); s += v[i]; }
  for (int off = 16; off; off >>= 1) s += __shfl_xor(s, off, 32);
  const float inv = 1.f / s;
  for (int i = 0; i < per; ++i) p[lane + (i << 5)] = v[i] * inv;
}

// ------------- split fused qkv [L*T,2048] -> q,k,v,g [L*4,T,32] ------------
// reference column order: col = d*16 + which*4 + h   (d<32, which<4, h<4)
__global__ void split_qkv_kernel(const float* qkv, float* q, float* k,
                                 float* v, float* g, int L, int tSh) {
  const long long idx = (long long)blockIdx.x * 256 + threadIdx.x;
  const int T = 1 << tSh;
  const long long total = ((long long)L << tSh) * 2048;
  if (idx >= total) return;
  const int col = (int)(idx & 2047);
  const long long lt = idx >> 11;
  const int t = (int)(lt & (T - 1));
  const int l = (int)(lt >> tSh);
  const int h = col & 3, which = (col >> 2) & 3, d = col >> 4;
  float* dst = which == 0 ? q : which == 1 ? k : which == 2 ? v : g;
  dst[((((long long)(l * 4 + h) << tSh) + t) << 5) + d] = qkv[idx];
}

// ------- ctx[l,t,h*32+d] = attn[(l*4+h),t,d] * sigmoid(g[(l*4+h),t,d]) -----
__global__ void gate_relayout_kernel(const float* attn, const float* g,
                                     float* ctx, int L, int tSh) {
  const long long idx = (long long)blockIdx.x * 256 + threadIdx.x;
  const int T = 1 << tSh;
  const long long total = ((long long)L * 4 << tSh) * 32;
  if (idx >= total) return;
  const int d = (int)(idx & 31);
  const long long r = idx >> 5;
  const int t = (int)(r & (T - 1));
  const long long lh = r >> tSh;
  const int h = (int)(lh & 3);
  const int l = (int)(lh >> 2);
  const float gv = g[idx];
  ctx[((((long long)l << tSh) + t) << 7) + h * 32 + d] =
      attn[idx] * (1.f / (1.f + __expf(-gv)));
}

__global__ void mul_sigmoid_kernel(const float* a, const float* b, float* o,
                                   long long n) {
  const long long i = (long long)blockIdx.x * 256 + threadIdx.x;
  if (i < n) { const float v = a[i] * b[i]; o[i] = 1.f / (1.f + __expf(-v)); }
}

__global__ void add2_kernel(const float* a, const float* b, float* o, long long n) {
  const long long i = (long long)blockIdx.x * 256 + threadIdx.x;
  if (i < n) o[i] = a[i] + b[i];
}

// ------------------------------ host helpers -------------------------------
static inline int blks(long long n, int b) { return (int)((n + b - 1) / b); }
static inline int ilog2(int v) { return __builtin_ctz((unsigned)v); }

static void run_gemm(hipStream_t st, const float* A, const float* B, float* C,
                     const float* bias, const float* residual, const float* gate,
                     int M, int N, int K, int nbatch,
                     long long ldaR, long long ldaKlo, long long ldaKhi, int kshA,
                     long long ldbN, long long ldbKlo, long long ldbKhi, int kshB,
                     long long ldcR, long long ldcN,
                     long long sA, long long sB, long long sC,
                     float alpha, int act) {
  GemmP p;
  p.A = A; p.B = B; p.C = C; p.bias = bias; p.residual = residual; p.gate = gate;
  p.M = M; p.N = N; p.K = K; p.act = act;
  p.kshA = kshA; p.kmaskA = (1 << kshA) - 1;
  p.kshB = kshB; p.kmaskB = (1 << kshB) - 1;
  p.ldaR = ldaR; p.ldaKlo = ldaKlo; p.ldaKhi = ldaKhi;
  p.ldbN = ldbN; p.ldbKlo = ldbKlo; p.ldbKhi = ldbKhi;
  p.ldcR = ldcR; p.ldcN = ldcN; p.sA = sA; p.sB = sB; p.sC = sC; p.alpha = alpha;
  dim3 grid((N + 63) / 64, (M + 63) / 64, nbatch);
  gemm_f16_wmma_kernel<<<grid, dim3(128, 1, 1), 0, st>>>(p);
}

// plain NN single-batch:  C[M,N] = A[M,K] . B[K,N]
static void run_gemm_nn(hipStream_t st, const float* A, const float* B, float* C,
                        const float* bias, const float* residual, const float* gate,
                        int M, int N, int K, float alpha, int act) {
  run_gemm(st, A, B, C, bias, residual, gate, M, N, K, 1,
           K, 1, 0, KSH_NONE, 1, N, 0, KSH_NONE, N, 1, 0, 0, 0, alpha, act);
}

static void run_small(hipStream_t st, const float* A, const float* B, float* C,
                      const float* bias, int M, int N, int K, int nbatch,
                      long long ldaR, long long ldaK, long long sA, long long sC) {
  SmallP p; p.A = A; p.B = B; p.C = C; p.bias = bias;
  p.M = M; p.N = N; p.K = K; p.ldaR = ldaR; p.ldaK = ldaK; p.sA = sA; p.sC = sC;
  dim3 grid(blks(M, 256), 1, nbatch);
  gemm_small_kernel<<<grid, dim3(256, 1, 1), 0, st>>>(p);
}

static void run_ln(hipStream_t st, const float* src, float* dst, const float* g,
                   const float* b, long long rows, int D) {
  ln_kernel<<<dim3(blks(rows, 8)), dim3(256), 0, st>>>(src, dst, g, b, rows, D);
}

// ---------------------------------------------------------------------------
// mhsa helper.  xln: LN'd input viewed per-batch: A(row,k)=xln[b*sA+row*ldaR+k]
// nb*Mb rows total = L*T.  Wqkv:[Cin,2048] W0:[128,Cin].
// Output written with strides (c_ldcR, c_ldcN, c_sC) + residual (same strides).
// ---------------------------------------------------------------------------
static void mhsa(hipStream_t st, const float* xln, int Cin,
                 int nb, int Mb, long long a_ldaR, long long a_sA,
                 int L, int T, const float* Wqkv, const float* W0,
                 const float* bias,
                 float* out, const float* res,
                 long long c_ldcR, long long c_ldcN, long long c_sC,
                 float* W_QKV, float* W_Q, float* W_K, float* W_V, float* W_G4,
                 float* W_SC, float* W_AT, float* W_CX) {
  const float scale = 0.17677669529663689f;  // 1/sqrt(32)
  const long long LT = (long long)L * T;
  const int tSh = ilog2(T);
  // qkv = xln @ Wqkv  -> W_QKV [L*T, 2048] (row order matches A batching)
  run_gemm(st, xln, Wqkv, W_QKV, nullptr, nullptr, nullptr,
           Mb, 2048, Cin, nb, a_ldaR, 1, 0, KSH_NONE, 1, 2048, 0, KSH_NONE,
           2048, 1, a_sA, 0, (long long)Mb * 2048, 1.f, 0);
  split_qkv_kernel<<<dim3(blks(LT * 2048, 256)), dim3(256), 0, st>>>(
      W_QKV, W_Q, W_K, W_V, W_G4, L, tSh);
  // scores = q.k^T * scale : batch = L*4
  run_gemm(st, W_Q, W_K, W_SC, nullptr, nullptr, nullptr,
           T, T, 32, L * 4, 32, 1, 0, KSH_NONE, 32, 1, 0, KSH_NONE,
           T, 1, (long long)T * 32, (long long)T * 32, (long long)T * T,
           scale, 0);
  softmax_bias_kernel<<<dim3(blks((long long)L * 4 * T, 8)), dim3(256), 0, st>>>(
      W_SC, bias, (long long)L * 4 * T, tSh, bias ? 4 : 1);
  // attn @ v
  run_gemm(st, W_SC, W_V, W_AT, nullptr, nullptr, nullptr,
           T, 32, T, L * 4, T, 1, 0, KSH_NONE, 1, 32, 0, KSH_NONE,
           32, 1, (long long)T * T, (long long)T * 32, (long long)T * 32,
           1.f, 0);
  gate_relayout_kernel<<<dim3(blks(LT * 128, 256)), dim3(256), 0, st>>>(
      W_AT, W_G4, W_CX, L, tSh);
  // out = res + ctx @ W0   (ctx rows ordered exactly like xln rows)
  run_gemm(st, W_CX, W0, out, nullptr, res, nullptr,
           Mb, Cin, 128, nb, 128, 1, 0, KSH_NONE, 1, Cin, 0, KSH_NONE,
           c_ldcR, c_ldcN, (long long)Mb * 128, 0, c_sC, 1.f, 0);
}

// tri_mult: out = sigmoid(z@lg)*(LN(einsum(a,b))@lz + b) + z
static void tri_mult(hipStream_t st, const float* z, void* const* d_in, int pb,
                     bool incoming, float* out,
                     float* W_T1, float* W_T2, float* W_TA, float* W_TB,
                     float* W_GT, float* W_ZZ, float* W_ZN) {
#define TP(i) ((const float*)d_in[(pb) + (i)])
  const long long n32 = 65536LL * 32;
  run_gemm_nn(st, z, TP(0), W_T1, TP(1), nullptr, nullptr, 65536, 32, 128, 1.f, 0);
  run_gemm_nn(st, z, TP(2), W_T2, TP(3), nullptr, nullptr, 65536, 32, 128, 1.f, 0);
  mul_sigmoid_kernel<<<dim3(blks(n32, 256)), dim3(256), 0, st>>>(W_T1, W_T2, W_TA, n32);
  run_gemm_nn(st, z, TP(4), W_T1, TP(5), nullptr, nullptr, 65536, 32, 128, 1.f, 0);
  run_gemm_nn(st, z, TP(6), W_T2, TP(7), nullptr, nullptr, 65536, 32, 128, 1.f, 0);
  mul_sigmoid_kernel<<<dim3(blks(n32, 256)), dim3(256), 0, st>>>(W_T1, W_T2, W_TB, n32);
  run_gemm_nn(st, z, TP(8), W_GT, TP(9), nullptr, nullptr, 65536, 128, 128, 1.f, 2);
  if (!incoming)  // zz[i,j,c] = sum_k a[i,k,c]*b[k,j,c], batch over c
    run_gemm(st, W_TA, W_TB, W_ZZ, nullptr, nullptr, nullptr,
             256, 256, 256, 32, 8192, 32, 0, KSH_NONE, 32, 8192, 0, KSH_NONE,
             8192, 32, 1, 1, 1, 1.f, 0);
  else            // zz[i,j,c] = sum_k a[k,i,c]*b[j,k,c]
    run_gemm(st, W_TA, W_TB, W_ZZ, nullptr, nullptr, nullptr,
             256, 256, 256, 32, 32, 8192, 0, KSH_NONE, 8192, 32, 0, KSH_NONE,
             8192, 32, 1, 1, 1, 1.f, 0);
  run_ln(st, W_ZZ, W_ZN, TP(12), TP(13), 65536, 32);
  run_gemm_nn(st, W_ZN, TP(10), out, TP(11), z, W_GT, 65536, 128, 32, 1.f, 0);
#undef TP
}

// ---------------------------------------------------------------------------
extern "C" void kernel_launch(void* const* d_in, const int* in_sizes, int n_in,
                              void* d_out, int out_size, void* d_ws, size_t ws_size,
                              hipStream_t stream) {
  (void)in_sizes; (void)n_in; (void)out_size; (void)ws_size;
#define P(i) ((const float*)d_in[i])
  const float* prw = P(0);   // [256,256,128]
  const float* msa = P(1);   // [128,256,256]
  float* out_prw = (float*)d_out;
  float* out_msa = out_prw + 8388608;

  float* w = (float*)d_ws;
  auto take = [&](long long n) { float* p = w; w += n; return p; };
  float* W_QKV = take(134217728LL);  // qkv staging
  float* W_Q  = take(8388608LL);
  float* W_K  = take(8388608LL);
  float* W_V  = take(8388608LL);
  float* W_G4 = take(8388608LL);
  float* W_SC = take(67108864LL);    // scores / FFN hidden / OPM opbuf
  float* W_AT = take(8388608LL);
  float* W_CX = take(8388608LL);
  float* W_X2 = take(8388608LL);
  float* W_LN = take(8388608LL);
  float* W_Z0 = take(8388608LL);
  float* W_Z1 = take(8388608LL);
  float* W_Z2 = take(8388608LL);
  float* W_T1 = take(2097152LL);
  float* W_T2 = take(2097152LL);
  float* W_TA = take(2097152LL);
  float* W_TB = take(2097152LL);
  float* W_GT = take(8388608LL);
  float* W_ZZ = take(2097152LL);
  float* W_ZN = take(2097152LL);
  float* W_BI = take(262144LL);
  float* W_Y1 = take(8388608LL);
  float* W_Y2 = take(8388608LL);
  float* W_Y3 = take(8388608LL);

  // ======================= MSA stack =======================
  run_ln(stream, msa, W_LN, P(2), P(3), 32768, 256);          // x1
  run_ln(stream, prw, W_Y1, P(4), P(5), 65536, 128);          // br
  run_small(stream, W_Y1, P(10), W_BI, P(11), 65536, 4, 128, 1, 128, 1, 0, 0);
  // row attention: lead S=128, T=256, Cin=256, contiguous view
  mhsa(stream, W_LN, 256, 1, 32768, 256, 0, 128, 256, P(8), P(9), W_BI,
       W_X2, W_LN, 256, 1, 0, W_QKV, W_Q, W_K, W_V, W_G4, W_SC, W_AT, W_CX);
  run_ln(stream, W_X2, W_LN, P(6), P(7), 32768, 256);         // x3
  // col attention: batch r=256, rows s=128, transposed view of x3; output
  // written back transposed into [s,r,c] with residual x3.
  mhsa(stream, W_LN, 256, 256, 128, 65536, 256, 256, 128, P(12), P(13), nullptr,
       W_X2, W_LN, 65536, 1, 256, W_QKV, W_Q, W_K, W_V, W_G4, W_SC, W_AT, W_CX);
  // FFN
  run_gemm_nn(stream, W_X2, P(14), W_SC, P(15), nullptr, nullptr,
              32768, 1024, 256, 1.f, 1);
  run_gemm_nn(stream, W_SC, P(16), out_msa, P(17), W_X2, nullptr,
              32768, 256, 1024, 1.f, 0);                      // msa_out

  // ==================== Outer product mean ====================
  run_ln(stream, out_msa, W_LN, P(18), P(19), 32768, 256);
  run_gemm_nn(stream, W_LN, P(20), W_T1, P(21), nullptr, nullptr,
              32768, 32, 256, 1.f, 0);                        // a [s, r*32+c]
  // op[(i,c),(j,d)] = (1/S) sum_s a[s,ic]*a[s,jd]  -> W_SC [8192,8192]
  run_gemm(stream, W_T1, W_T1, W_SC, nullptr, nullptr, nullptr,
           8192, 8192, 128, 1, 1, 8192, 0, KSH_NONE, 1, 8192, 0, KSH_NONE,
           8192, 1, 0, 0, 0, 1.f / 128.f, 0);
  // z0[i,j,:] = op[i,j,(c,d)] @ fc2 + b + prw  (K gather: c=k>>5, d=k&31)
  run_gemm(stream, W_SC, P(22), W_Z0, P(23), prw, nullptr,
           256, 128, 1024, 256, 32, 1, 8192, 5, 1, 128, 0, KSH_NONE,
           128, 1, 262144, 0, 32768, 1.f, 0);

  // ==================== Triangle multiplications ====================
  tri_mult(stream, W_Z0, d_in, 24, false, W_Z1, W_T1, W_T2, W_TA, W_TB,
           W_GT, W_ZZ, W_ZN);
  tri_mult(stream, W_Z1, d_in, 38, true, W_Z2, W_T1, W_T2, W_TA, W_TB,
           W_GT, W_ZZ, W_ZN);

  // ======================= Pair stack =======================
  run_ln(stream, W_Z2, W_LN, P(52), P(53), 65536, 128);       // y1
  run_small(stream, W_LN, P(58), W_BI, P(59), 65536, 4, 128, 1, 128, 1, 0, 0);
  // start attention (lead i, T=256)
  mhsa(stream, W_LN, 128, 1, 65536, 128, 0, 256, 256, P(56), P(57), W_BI,
       W_Y1, W_LN, 128, 1, 0, W_QKV, W_Q, W_K, W_V, W_G4, W_SC, W_AT, W_CX);
  run_ln(stream, W_Y1, W_LN, P(54), P(55), 65536, 128);       // y3
  // end-bias on transposed view: proj[a,b,h] from y3[b,a,:]
  run_small(stream, W_LN, P(62), W_BI, P(63), 256, 4, 128, 256, 32768, 1, 128, 1024);
  // end attention: batch a=256, rows b=256, transposed view; output written
  // back transposed with residual y3.
  mhsa(stream, W_LN, 128, 256, 256, 32768, 128, 256, 256, P(60), P(61), W_BI,
       W_Y2, W_LN, 32768, 1, 128, W_QKV, W_Q, W_K, W_V, W_G4, W_SC, W_AT, W_CX);
  // FFN
  run_gemm_nn(stream, W_Y2, P(64), W_SC, P(65), nullptr, nullptr,
              65536, 512, 128, 1.f, 1);
  run_gemm_nn(stream, W_SC, P(66), W_Y3, P(67), W_Y2, nullptr,
              65536, 128, 512, 1.f, 0);
  // prw_out = pair_stack(x) + x
  add2_kernel<<<dim3(blks(8388608LL, 256)), dim3(256), 0, stream>>>(
      W_Y3, W_Z2, out_prw, 8388608LL);
#undef P
}
